// EncoderHistoryInformation_60473139527930
// MI455X (gfx1250) — matrix operand order repaired and checked by reading, not compiled, not run on hardware
//
#include <hip/hip_runtime.h>

// ---------------------------------------------------------------------------
// LSTM (Keras cell, gates i,f,g,o) over T=50 steps.
// B=2048, T=50, F=256, H=256, 4H=1024.
// bf16 WMMA (f32 accum). Each 256-thread block owns 16 batch rows for the
// whole recurrence; weights pre-converted/transposed to bf16 in d_ws.
// K-loops kept as rolled loops (unroll 2) to stay under 256 VGPRs: no
// scratch spills, no VGPR-MSB mode, multiple waves/SIMD for latency hiding.
// ---------------------------------------------------------------------------

typedef __bf16 bf16_t;
typedef __attribute__((ext_vector_type(16))) __bf16 v16bf;
typedef __attribute__((ext_vector_type(8)))  __bf16 v8bf;
typedef __attribute__((ext_vector_type(8)))  float  v8f;
typedef __attribute__((ext_vector_type(4)))  float  v4f;

#define LSTM_B   2048
#define LSTM_T   50
#define LSTM_F   256
#define LSTM_H   256
#define LSTM_4H  1024
#define BM       16            // batch rows per block
#define LDS_PAD  8             // bf16 elems of row padding (16B) -> bank spread
#define GATE_STRIDE (16 * 16 * 256)   // elements between gate column groups

__device__ __forceinline__ float sigmoid_f(float x) {
  x = fminf(fmaxf(x, -20.0f), 20.0f);
  return 1.0f / (1.0f + __expf(-x));
}
__device__ __forceinline__ float tanh_f(float x) {
  x = fminf(fmaxf(x, -15.0f), 15.0f);
  float e = __expf(2.0f * x);
  return (e - 1.0f) / (e + 1.0f);
}

// A-fragment gather: two contiguous 16B LDS chunks (K 0..7 and K 8..15 of
// this lane's half-wave 16-K window), matching the contiguous B K-run.
__device__ __forceinline__ v16bf load_a_frag(const bf16_t* ap) {
  v8bf lo = *(const v8bf*)ap;
  v8bf hi = *(const v8bf*)(ap + 8);
  return __builtin_shufflevector(lo, hi,
      0, 1, 2, 3, 4, 5, 6, 7, 8, 9, 10, 11, 12, 13, 14, 15);
}

__device__ __forceinline__ v8f wmma_acc(v16bf a, const bf16_t* bp, v8f c) {
  v16bf b = *(const v16bf*)bp;   // 16 contiguous K values (32B, 2x b128)
  return __builtin_amdgcn_wmma_f32_16x16x32_bf16(
      false, a, false, b, (short)0, c, false, false);
}

// Pre-kernel: W[F][4H], U[H][4H] (f32, row-major) -> Wt[4H][F], Ut[4H][H]
// (bf16, n-major) so B-fragment K-runs are contiguous in memory.
__global__ __launch_bounds__(256) void lstm_prep_kernel(
    const float* __restrict__ W, const float* __restrict__ U,
    bf16_t* __restrict__ Wt, bf16_t* __restrict__ Ut) {
  int idx = blockIdx.x * blockDim.x + threadIdx.x;   // 0 .. 4H*F-1
  int n = idx >> 8;        // 0..1023
  int k = idx & 255;       // 0..255
  Wt[idx] = (bf16_t)W[(size_t)k * LSTM_4H + n];
  Ut[idx] = (bf16_t)U[(size_t)k * LSTM_4H + n];
}

__global__ __launch_bounds__(256) void lstm_wmma_kernel(
    const float*  __restrict__ x,     // [B, T, F]
    const float*  __restrict__ bias,  // [4H]
    const bf16_t* __restrict__ Wt,    // [4H, F] n-major bf16
    const bf16_t* __restrict__ Ut,    // [4H, H] n-major bf16
    float*        __restrict__ out)   // [2, B, H] : h then c
{
  __shared__ bf16_t xa[BM][LSTM_F + LDS_PAD];  // x_t tile, bf16 row-major
  __shared__ bf16_t ha[BM][LSTM_H + LDS_PAD];  // h tile,   bf16 row-major

  const int tid  = threadIdx.x;
  const int lane = tid & 31;
  const int wave = tid >> 5;          // 0..7
  const int row0 = blockIdx.x * BM;   // batch base
  const int lhi  = lane >> 4;         // 0/1: half-wave
  const int llo  = lane & 15;

  // zero h staging (h0 = 0)
  for (int i = tid; i < BM * (LSTM_H + LDS_PAD); i += 256)
    ((bf16_t*)ha)[i] = (bf16_t)0.0f;

  // Bias per owned tile: tiles are (gate g, column-block cb) with
  // cb in {wave, wave+8}; z column = (g*16+cb)*16 + llo.
  float biasv[8];
#pragma unroll
  for (int g = 0; g < 4; ++g) {
#pragma unroll
    for (int p = 0; p < 2; ++p)
      biasv[g * 2 + p] = bias[((g * 16 + wave + p * 8) << 4) + llo];
  }

  // Persistent recurrent state in WMMA C-layout (M = r + 8*lhi, N = llo).
  v8f c0 = {0.f,0.f,0.f,0.f,0.f,0.f,0.f,0.f};
  v8f c1 = c0, h0 = c0, h1 = c0;

  // cooperative x loader coordinates
  const int xr = tid >> 4;            // row 0..15
  const int xc = (tid & 15) << 4;     // col base, 16 floats

  // Weight base pointers for this lane's two column blocks (p = 0, 1):
  // row (n) index = (cb*16 + llo), K offset starts at 16*lhi.
  const bf16_t* wb[2];
  const bf16_t* ub[2];
#pragma unroll
  for (int p = 0; p < 2; ++p) {
    const int cb = wave + p * 8;
    wb[p] = Wt + (size_t)((cb << 4) + llo) * LSTM_F + 16 * lhi;
    ub[p] = Ut + (size_t)((cb << 4) + llo) * LSTM_H + 16 * lhi;
  }

  for (int t = 0; t < LSTM_T; ++t) {
    // ---- stage x_t tile into LDS as bf16 -------------------------------
    {
      const float* src = x + ((size_t)(row0 + xr) * LSTM_T + t) * LSTM_F + xc;
      v4f f0 = *(const v4f*)(src + 0);
      v4f f1 = *(const v4f*)(src + 4);
      v4f f2 = *(const v4f*)(src + 8);
      v4f f3 = *(const v4f*)(src + 12);
      v8bf p0, p1;
#pragma unroll
      for (int j = 0; j < 4; ++j) { p0[j] = (bf16_t)f0[j]; p0[4 + j] = (bf16_t)f1[j]; }
#pragma unroll
      for (int j = 0; j < 4; ++j) { p1[j] = (bf16_t)f2[j]; p1[4 + j] = (bf16_t)f3[j]; }
      *(v8bf*)&xa[xr][xc]     = p0;
      *(v8bf*)&xa[xr][xc + 8] = p1;
      if (t + 1 < LSTM_T) __builtin_prefetch(src + LSTM_F, 0, 0);
    }
    __syncthreads();   // barrier A: x_t & h visible, prior reads complete

    // ---- two column-block passes: cb = wave, wave+8 --------------------
#pragma unroll
    for (int p = 0; p < 2; ++p) {
      v8f a0, a1, a2, a3;                 // i, f, g, o accumulators
#pragma unroll
      for (int r = 0; r < 8; ++r) {
        a0[r] = biasv[0 + p];
        a1[r] = biasv[2 + p];
        a2[r] = biasv[4 + p];
        a3[r] = biasv[6 + p];
      }
      const bf16_t* wp = wb[p];
      const bf16_t* up = ub[p];

      // z += x_t @ W   (A from xa LDS, B streamed from L2-resident Wt)
#pragma unroll 2
      for (int kc = 0; kc < 8; ++kc) {
        v16bf a = load_a_frag(&xa[llo][kc * 32 + 16 * lhi]);
        const bf16_t* bp = wp + kc * 32;
        a0 = wmma_acc(a, bp + 0 * GATE_STRIDE, a0);
        a1 = wmma_acc(a, bp + 1 * GATE_STRIDE, a1);
        a2 = wmma_acc(a, bp + 2 * GATE_STRIDE, a2);
        a3 = wmma_acc(a, bp + 3 * GATE_STRIDE, a3);
      }
      // z += h @ U
#pragma unroll 2
      for (int kc = 0; kc < 8; ++kc) {
        v16bf a = load_a_frag(&ha[llo][kc * 32 + 16 * lhi]);
        const bf16_t* bp = up + kc * 32;
        a0 = wmma_acc(a, bp + 0 * GATE_STRIDE, a0);
        a1 = wmma_acc(a, bp + 1 * GATE_STRIDE, a1);
        a2 = wmma_acc(a, bp + 2 * GATE_STRIDE, a2);
        a3 = wmma_acc(a, bp + 3 * GATE_STRIDE, a3);
      }

      // gate fusion, in-register recurrent state update
      v8f& c = p ? c1 : c0;
      v8f& h = p ? h1 : h0;
#pragma unroll
      for (int r = 0; r < 8; ++r) {
        float iv = sigmoid_f(a0[r]);
        float fv = sigmoid_f(a1[r]);
        float gv = tanh_f(a2[r]);
        float ov = sigmoid_f(a3[r]);
        float cn = fv * c[r] + iv * gv;
        c[r] = cn;
        h[r] = ov * tanh_f(cn);
      }
    }

    __syncthreads();   // barrier B: all LDS reads of this step done

    // ---- publish h_new (bf16) for next step's h @ U --------------------
#pragma unroll
    for (int p = 0; p < 2; ++p) {
      const int cb = wave + p * 8;
      const v8f& h = p ? h1 : h0;
#pragma unroll
      for (int r = 0; r < 8; ++r)
        ha[r + 8 * lhi][(cb << 4) + llo] = (bf16_t)h[r];
    }
  }

  // ---- final h, c -> out (f32): out[0:B*H]=h, out[B*H:2BH]=c ----------
#pragma unroll
  for (int p = 0; p < 2; ++p) {
    const int cb = wave + p * 8;
    const v8f& h = p ? h1 : h0;
    const v8f& c = p ? c1 : c0;
#pragma unroll
    for (int r = 0; r < 8; ++r) {
      const int m   = r + 8 * lhi;
      const int col = (cb << 4) + llo;
      out[(size_t)(row0 + m) * LSTM_H + col] = h[r];
      out[(size_t)LSTM_B * LSTM_H + (size_t)(row0 + m) * LSTM_H + col] = c[r];
    }
  }
}

extern "C" void kernel_launch(void* const* d_in, const int* in_sizes, int n_in,
                              void* d_out, int out_size, void* d_ws, size_t ws_size,
                              hipStream_t stream) {
  (void)in_sizes; (void)n_in; (void)out_size; (void)ws_size;
  const float* x    = (const float*)d_in[0];   // [B, T, F]
  const float* W    = (const float*)d_in[1];   // [F, 4H]
  const float* U    = (const float*)d_in[2];   // [H, 4H]
  const float* bias = (const float*)d_in[3];   // [4H]

  bf16_t* Wt = (bf16_t*)d_ws;                        // [4H, F] bf16 = 512 KB
  bf16_t* Ut = Wt + (size_t)LSTM_4H * LSTM_F;        // [4H, H] bf16 = 512 KB

  lstm_prep_kernel<<<(LSTM_4H * LSTM_F) / 256, 256, 0, stream>>>(W, U, Wt, Ut);
  lstm_wmma_kernel<<<LSTM_B / BM, 256, 0, stream>>>(x, bias, Wt, Ut, (float*)d_out);
}
